// Linear_75874892251182
// MI455X (gfx1250) — compile-verified, matrix operand order checked by reading, not run on hardware
//
#include <hip/hip_runtime.h>
#include <stdint.h>

// ---- types ----
typedef __attribute__((ext_vector_type(16))) __bf16 bf16x16;
typedef __attribute__((ext_vector_type(8)))  __bf16 bf16x8;
typedef __attribute__((ext_vector_type(4)))  __bf16 bf16x4;
typedef __attribute__((ext_vector_type(8)))  float  v8f;

#define IN_F  4096
#define OUT_F 4096
#define N_TOK 8192

// pass-2 GEMM tiling
#define BM 128
#define BN 128
#define BK 64
#define LDH 72      // halfword stride: 64 + 8 pad (144B rows; 16B aligned, bank-balanced)
#define TILE_HW (BM * LDH)

#define USE_ASYNC_COPY 1

// ==========================================================================
// Pass 1a: x (fp32) -> x_hi + x_lo (bf16 split, ~fp32-accurate recomposition)
// ==========================================================================
__global__ __launch_bounds__(256)
void convert_x_kernel(const float* __restrict__ x,
                      __bf16* __restrict__ xh,
                      __bf16* __restrict__ xl, int n4) {
    int i0 = blockIdx.x * blockDim.x + threadIdx.x;
    int stride = gridDim.x * blockDim.x;
    for (int i = i0; i < n4; i += stride) {
        float4 v = ((const float4*)x)[i];
        float a[4] = {v.x, v.y, v.z, v.w};
        bf16x4 hi, lo;
        #pragma unroll
        for (int e = 0; e < 4; ++e) {
            __bf16 h = (__bf16)a[e];           // RNE truncation
            hi[e] = h;
            lo[e] = (__bf16)(a[e] - (float)h); // residual
        }
        ((bf16x4*)xh)[i] = hi;
        ((bf16x4*)xl)[i] = lo;
    }
}

// ==========================================================================
// Pass 1b: ternary weights W = (wpos>0) - (wneg>0) in {-1,0,1}, exact in bf16
// ==========================================================================
__global__ __launch_bounds__(256)
void convert_w_kernel(const float* __restrict__ wp,
                      const float* __restrict__ wn,
                      __bf16* __restrict__ wt, int n4) {
    int i0 = blockIdx.x * blockDim.x + threadIdx.x;
    int stride = gridDim.x * blockDim.x;
    for (int i = i0; i < n4; i += stride) {
        float4 p4 = ((const float4*)wp)[i];
        float4 q4 = ((const float4*)wn)[i];
        float p[4] = {p4.x, p4.y, p4.z, p4.w};
        float q[4] = {q4.x, q4.y, q4.z, q4.w};
        bf16x4 t;
        #pragma unroll
        for (int e = 0; e < 4; ++e)
            t[e] = (__bf16)((p[e] > 0.0f ? 1.0f : 0.0f) - (q[e] > 0.0f ? 1.0f : 0.0f));
        ((bf16x4*)wt)[i] = t;
    }
}

// ==========================================================================
// Pass 2: bf16 WMMA GEMM with DOUBLE-BUFFERED async global->LDS staging.
// y[t,o] = sum_k (xh+xl)[t,k] * W[o,k]
// 256 thr = 8 waves (4x2), each wave 32x64 of a 128x128 C tile.
// Per stage, each wave issues 12 GLOBAL_LOAD_ASYNC_TO_LDS_B128 (ASYNCcnt);
// async loads retire in order, so s_wait_asynccnt 12 == "previous stage done".
// ==========================================================================
#if USE_ASYNC_COPY
#define ASYNC_B128(ldsoff, base, voff)                                 \
    asm volatile("global_load_async_to_lds_b128 %0, %1, %2"            \
                 :: "v"(ldsoff), "v"(voff), "s"(base) : "memory")
#endif

__device__ __forceinline__ void stage_tiles(
    const __bf16* __restrict__ xh, const __bf16* __restrict__ xl,
    const __bf16* __restrict__ wt,
    int m_tile, int n_tile, int kc, int tid,
    uint32_t hi_base, uint32_t lo_base, uint32_t bs_base,
    __bf16* hi_p, __bf16* lo_p, __bf16* bs_p) {
    #pragma unroll
    for (int it = 0; it < 4; ++it) {
        int f  = tid + 256 * it;      // 0..1023 chunks of 8 halfwords
        int r  = f >> 3;              // row 0..127
        int c8 = f & 7;               // chunk 0..7
        int gx = ((m_tile + r) * IN_F + kc + c8 * 8) * 2;   // byte offsets (fit int32)
        int gw = ((n_tile + r) * IN_F + kc + c8 * 8) * 2;
        uint32_t loff = (uint32_t)((r * LDH + c8 * 8) * 2);
#if USE_ASYNC_COPY
        ASYNC_B128(hi_base + loff, xh, gx);
        ASYNC_B128(lo_base + loff, xl, gx);
        ASYNC_B128(bs_base + loff, wt, gw);
#else
        *(bf16x8*)(hi_p + r * LDH + c8 * 8) = *(const bf16x8*)((const char*)xh + gx);
        *(bf16x8*)(lo_p + r * LDH + c8 * 8) = *(const bf16x8*)((const char*)xl + gx);
        *(bf16x8*)(bs_p + r * LDH + c8 * 8) = *(const bf16x8*)((const char*)wt + gw);
#endif
    }
}

__global__ __launch_bounds__(256)
void gemm_bf16_wmma(const __bf16* __restrict__ xh,
                    const __bf16* __restrict__ xl,
                    const __bf16* __restrict__ wt,
                    float* __restrict__ out) {
    __shared__ __bf16 As_hi[2][TILE_HW];
    __shared__ __bf16 As_lo[2][TILE_HW];
    __shared__ __bf16 Bs[2][TILE_HW];

    const int tid  = threadIdx.x;
    const int lane = tid & 31;
    const int wave = tid >> 5;
    const int wm   = wave >> 1;     // 0..3 : 32-row band
    const int wno  = wave & 1;      // 0..1 : 64-col band
    const int half = lane >> 4;
    const int nl   = lane & 15;
    const int k0   = half * 8;      // halfword offset of this half-wave's first K run

    const int n_tile = blockIdx.x * BN;
    const int m_tile = blockIdx.y * BM;

    uint32_t hi_base[2], lo_base[2], bs_base[2];
    #pragma unroll
    for (int b = 0; b < 2; ++b) {
        hi_base[b] = (uint32_t)(uintptr_t)&As_hi[b][0];
        lo_base[b] = (uint32_t)(uintptr_t)&As_lo[b][0];
        bs_base[b] = (uint32_t)(uintptr_t)&Bs[b][0];
    }

    v8f acc[2][4] = {};

    const int nk = IN_F / BK;   // 64 K-chunks

    // prologue: stage chunk 0 into buffer 0
    stage_tiles(xh, xl, wt, m_tile, n_tile, 0, tid,
                hi_base[0], lo_base[0], bs_base[0],
                &As_hi[0][0], &As_lo[0][0], &Bs[0][0]);

    for (int i = 0; i < nk; ++i) {
        const int cur = i & 1;
        if (i + 1 < nk) {
            // issue next stage into the other buffer BEFORE waiting on this one
            stage_tiles(xh, xl, wt, m_tile, n_tile, (i + 1) * BK, tid,
                        hi_base[cur ^ 1], lo_base[cur ^ 1], bs_base[cur ^ 1],
                        &As_hi[cur ^ 1][0], &As_lo[cur ^ 1][0], &Bs[cur ^ 1][0]);
#if USE_ASYNC_COPY
            // 12 newer ops may stay in flight; the 12 of stage i must retire
            asm volatile("s_wait_asynccnt 0xc" ::: "memory");
#endif
        } else {
#if USE_ASYNC_COPY
            asm volatile("s_wait_asynccnt 0x0" ::: "memory");
#endif
        }
        __syncthreads();

        const __bf16* hs = &As_hi[cur][0];
        const __bf16* ls = &As_lo[cur][0];
        const __bf16* bs = &Bs[cur][0];

        // ---------- two WMMA K-steps per staged tile ----------
        #pragma unroll
        for (int kk = 0; kk < 2; ++kk) {
            const int kb = kk * 32 + k0;
            // lane<16: K kb+0..7 and kb+16..23 ; lane>=16: +8..15 and +24..31
            bf16x16 a_hi[2], a_lo[2], bfr[4];
            #pragma unroll
            for (int mi = 0; mi < 2; ++mi) {
                int row = wm * 32 + mi * 16 + nl;
                const __bf16* ph = hs + row * LDH + kb;
                const __bf16* pl = ls + row * LDH + kb;
                union { bf16x16 v; bf16x8 h[2]; } uh, ul;
                uh.h[0] = *(const bf16x8*)(ph);
                uh.h[1] = *(const bf16x8*)(ph + 16);
                ul.h[0] = *(const bf16x8*)(pl);
                ul.h[1] = *(const bf16x8*)(pl + 16);
                a_hi[mi] = uh.v;
                a_lo[mi] = ul.v;
            }
            #pragma unroll
            for (int ni = 0; ni < 4; ++ni) {
                int row = wno * 64 + ni * 16 + nl;
                const __bf16* pb = bs + row * LDH + kb;
                union { bf16x16 v; bf16x8 h[2]; } ub;
                ub.h[0] = *(const bf16x8*)(pb);
                ub.h[1] = *(const bf16x8*)(pb + 16);
                bfr[ni] = ub.v;
            }
            #pragma unroll
            for (int mi = 0; mi < 2; ++mi)
                #pragma unroll
                for (int ni = 0; ni < 4; ++ni) {
                    acc[mi][ni] = __builtin_amdgcn_wmma_f32_16x16x32_bf16(
                        false, a_hi[mi], false, bfr[ni], (short)0, acc[mi][ni], false, false);
                    acc[mi][ni] = __builtin_amdgcn_wmma_f32_16x16x32_bf16(
                        false, a_lo[mi], false, bfr[ni], (short)0, acc[mi][ni], false, false);
                }
        }
        // protects buffer `cur` from stage i+2 async writes issued next iteration
        __syncthreads();
    }

    // ---------- write C ----------
    #pragma unroll
    for (int mi = 0; mi < 2; ++mi)
        #pragma unroll
        for (int ni = 0; ni < 4; ++ni) {
            int mb = m_tile + wm * 32 + mi * 16 + half * 8;
            int n  = n_tile + wno * 64 + ni * 16 + nl;
            float* o = out + (size_t)mb * OUT_F + n;
            #pragma unroll
            for (int r = 0; r < 8; ++r)
                o[(size_t)r * OUT_F] = acc[mi][ni][r];
        }
}

// ==========================================================================
// Fallback: fused single-pass kernel (workspace-free)
// ==========================================================================
__device__ __forceinline__ __bf16 f32_to_bf16_bits(float f) {
    union { float f; uint32_t u; } in; in.f = f;
    uint32_t u = in.u;
    u += 0x7FFFu + ((u >> 16) & 1u);
    union { uint16_t u; __bf16 b; } o; o.u = (uint16_t)(u >> 16);
    return o.b;
}
__device__ __forceinline__ float bf16_to_f32_bits(__bf16 b) {
    union { __bf16 b; uint16_t u; } in; in.b = b;
    union { uint32_t u; float f; } o; o.u = ((uint32_t)in.u) << 16;
    return o.f;
}

__global__ __launch_bounds__(256)
void ternary_linear_fused(const float* __restrict__ x,
                          const float* __restrict__ wpos,
                          const float* __restrict__ wneg,
                          float* __restrict__ out) {
    const int FLDH = 40;
    __shared__ __bf16 As_hi[128 * 40];
    __shared__ __bf16 As_lo[128 * 40];
    __shared__ __bf16 Bsf[128 * 40];

    const int tid  = threadIdx.x;
    const int lane = tid & 31;
    const int wave = tid >> 5;
    const int wm   = wave >> 1;
    const int wno  = wave & 1;
    const int half = lane >> 4;
    const int nl   = lane & 15;
    const int k0   = half * 8;

    const int n_tile = blockIdx.x * 128;
    const int m_tile = blockIdx.y * 128;

    v8f acc[2][4] = {};

    for (int kc = 0; kc < IN_F; kc += 32) {
        #pragma unroll
        for (int it = 0; it < 4; ++it) {
            int f  = tid + 256 * it;
            int r  = f >> 3;
            int c4 = f & 7;
            const float4 xv = *(const float4*)(x + (size_t)(m_tile + r) * IN_F + kc + c4 * 4);
            float v[4] = {xv.x, xv.y, xv.z, xv.w};
            bf16x4 hi, lo;
            #pragma unroll
            for (int e = 0; e < 4; ++e) {
                __bf16 h = f32_to_bf16_bits(v[e]);
                hi[e] = h;
                lo[e] = f32_to_bf16_bits(v[e] - bf16_to_f32_bits(h));
            }
            *(bf16x4*)(As_hi + r * FLDH + c4 * 4) = hi;
            *(bf16x4*)(As_lo + r * FLDH + c4 * 4) = lo;
            const float4 pv = *(const float4*)(wpos + (size_t)(n_tile + r) * IN_F + kc + c4 * 4);
            const float4 nv = *(const float4*)(wneg + (size_t)(n_tile + r) * IN_F + kc + c4 * 4);
            float p[4] = {pv.x, pv.y, pv.z, pv.w};
            float q[4] = {nv.x, nv.y, nv.z, nv.w};
            bf16x4 t;
            #pragma unroll
            for (int e = 0; e < 4; ++e)
                t[e] = f32_to_bf16_bits((p[e] > 0.0f ? 1.0f : 0.0f) - (q[e] > 0.0f ? 1.0f : 0.0f));
            *(bf16x4*)(Bsf + r * FLDH + c4 * 4) = t;
        }
        __syncthreads();

        bf16x16 a_hi[2], a_lo[2], bfr[4];
        #pragma unroll
        for (int mi = 0; mi < 2; ++mi) {
            int row = wm * 32 + mi * 16 + nl;
            const __bf16* ph = As_hi + row * FLDH + k0;
            const __bf16* pl = As_lo + row * FLDH + k0;
            union { bf16x16 v; bf16x8 h[2]; } uh, ul;
            uh.h[0] = *(const bf16x8*)(ph);
            uh.h[1] = *(const bf16x8*)(ph + 16);
            ul.h[0] = *(const bf16x8*)(pl);
            ul.h[1] = *(const bf16x8*)(pl + 16);
            a_hi[mi] = uh.v; a_lo[mi] = ul.v;
        }
        #pragma unroll
        for (int ni = 0; ni < 4; ++ni) {
            int row = wno * 64 + ni * 16 + nl;
            const __bf16* pb = Bsf + row * FLDH + k0;
            union { bf16x16 v; bf16x8 h[2]; } ub;
            ub.h[0] = *(const bf16x8*)(pb);
            ub.h[1] = *(const bf16x8*)(pb + 16);
            bfr[ni] = ub.v;
        }
        #pragma unroll
        for (int mi = 0; mi < 2; ++mi)
            #pragma unroll
            for (int ni = 0; ni < 4; ++ni) {
                acc[mi][ni] = __builtin_amdgcn_wmma_f32_16x16x32_bf16(
                    false, a_hi[mi], false, bfr[ni], (short)0, acc[mi][ni], false, false);
                acc[mi][ni] = __builtin_amdgcn_wmma_f32_16x16x32_bf16(
                    false, a_lo[mi], false, bfr[ni], (short)0, acc[mi][ni], false, false);
            }
        __syncthreads();
    }

    #pragma unroll
    for (int mi = 0; mi < 2; ++mi)
        #pragma unroll
        for (int ni = 0; ni < 4; ++ni) {
            int mb = m_tile + wm * 32 + mi * 16 + half * 8;
            int n  = n_tile + wno * 64 + ni * 16 + nl;
            float* o = out + (size_t)mb * OUT_F + n;
            #pragma unroll
            for (int r = 0; r < 8; ++r)
                o[(size_t)r * OUT_F] = acc[mi][ni][r];
        }
}

// ==========================================================================
extern "C" void kernel_launch(void* const* d_in, const int* in_sizes, int n_in,
                              void* d_out, int out_size, void* d_ws, size_t ws_size,
                              hipStream_t stream) {
    const float* x  = (const float*)d_in[0];
    const float* wp = (const float*)d_in[1];
    const float* wn = (const float*)d_in[2];
    float* out = (float*)d_out;

    const size_t x_elems = (size_t)N_TOK * IN_F;    // 33.5M
    const size_t w_elems = (size_t)OUT_F * IN_F;    // 16.8M
    const size_t need = (2 * x_elems + w_elems) * sizeof(__bf16);  // 160 MB

    if (ws_size >= need) {
        __bf16* xh = (__bf16*)d_ws;
        __bf16* xl = xh + x_elems;
        __bf16* wt = xl + x_elems;
        hipLaunchKernelGGL(convert_x_kernel, dim3(4096), dim3(256), 0, stream,
                           x, xh, xl, (int)(x_elems / 4));
        hipLaunchKernelGGL(convert_w_kernel, dim3(4096), dim3(256), 0, stream,
                           wp, wn, wt, (int)(w_elems / 4));
        hipLaunchKernelGGL(gemm_bf16_wmma, dim3(OUT_F / BN, N_TOK / BM), dim3(256), 0, stream,
                           xh, xl, wt, out);
    } else {
        hipLaunchKernelGGL(ternary_linear_fused, dim3(OUT_F / 128, N_TOK / 128), dim3(256), 0, stream,
                           x, wp, wn, out);
    }
}
